// TopKSAE_42855183679657
// MI455X (gfx1250) — compile-verified, matrix operand order checked by reading, not run on hardware
//
#include <hip/hip_runtime.h>
#include <hip/hip_bf16.h>

// ---------------------------------------------------------------------------
// TopK SAE forward for MI455X (gfx1250, wave32, WMMA).
//   encode:  pre = x @ W_enc^T + b_enc          (bf16 WMMA, f32 accum, ping-pong prefetch)
//   topk:    exact per-row radix-select (k=32)  (row staged in 96KB dynamic LDS)
//   decode:  recon = latents @ W_dec^T          (sparse gather on W_dec^T, coalesced)
// d_out = [recon 4096x768 | latents 4096x24576 | aux_loss(1)]  (fp32)
// Workspace (~120.6 MB): x_bf16 | W_enc_bf16 | W_decT | vals | idx
// ---------------------------------------------------------------------------

typedef __bf16 bf16;
typedef bf16  v16bf __attribute__((ext_vector_type(16)));
typedef bf16  v8bf  __attribute__((ext_vector_type(8)));
typedef float v8f   __attribute__((ext_vector_type(8)));
typedef float v4f   __attribute__((ext_vector_type(4)));

#define D_MODEL 768
#define D_SAE   24576
#define KTOP    32
#define BATCH   4096

// ---------------------------- fp32 -> bf16 convert -------------------------
__global__ __launch_bounds__(256)
void cvt_bf16_kernel(const float* __restrict__ in, bf16* __restrict__ out, int n8) {
    int i = blockIdx.x * 256 + threadIdx.x;
    if (i >= n8) return;
    const v4f* p = (const v4f*)in + (size_t)i * 2;
    v4f lo = p[0], hi = p[1];
    v8bf o;
    o[0] = (bf16)lo[0]; o[1] = (bf16)lo[1]; o[2] = (bf16)lo[2]; o[3] = (bf16)lo[3];
    o[4] = (bf16)hi[0]; o[5] = (bf16)hi[1]; o[6] = (bf16)hi[2]; o[7] = (bf16)hi[3];
    ((v8bf*)out)[i] = o;
}

// ------------------------ encode GEMM (bf16 WMMA) --------------------------
// pre[b,s] = sum_d x[b,d]*W_enc[s,d] + b_enc[s]
// Block tile: M=64 x N=128.  8 waves = 4(M) x 2(N); wave tile 16x64 = 4 accums.
// K-loop double-buffered; unroll forced to 1 so the ping-pong register sets
// survive regalloc and WMMAs overlap the next fragment loads (partial waits).
struct Frags { v16bf a, b0, b1, b2, b3; };

__device__ __forceinline__ void load_frags(Frags& f,
        const bf16* __restrict__ xrow, const bf16* __restrict__ w0,
        const bf16* __restrict__ w1,   const bf16* __restrict__ w2,
        const bf16* __restrict__ w3,   int ka) {
    const int kb = ka + 16;
    ((v8bf*)&f.a )[0] = *(const v8bf*)(xrow + ka); ((v8bf*)&f.a )[1] = *(const v8bf*)(xrow + kb);
    ((v8bf*)&f.b0)[0] = *(const v8bf*)(w0   + ka); ((v8bf*)&f.b0)[1] = *(const v8bf*)(w0   + kb);
    ((v8bf*)&f.b1)[0] = *(const v8bf*)(w1   + ka); ((v8bf*)&f.b1)[1] = *(const v8bf*)(w1   + kb);
    ((v8bf*)&f.b2)[0] = *(const v8bf*)(w2   + ka); ((v8bf*)&f.b2)[1] = *(const v8bf*)(w2   + kb);
    ((v8bf*)&f.b3)[0] = *(const v8bf*)(w3   + ka); ((v8bf*)&f.b3)[1] = *(const v8bf*)(w3   + kb);
}

#define WMMA4(F)                                                                                 \
    acc0 = __builtin_amdgcn_wmma_f32_16x16x32_bf16(false, (F).a, false, (F).b0, (short)0, acc0, false, false); \
    acc1 = __builtin_amdgcn_wmma_f32_16x16x32_bf16(false, (F).a, false, (F).b1, (short)0, acc1, false, false); \
    acc2 = __builtin_amdgcn_wmma_f32_16x16x32_bf16(false, (F).a, false, (F).b2, (short)0, acc2, false, false); \
    acc3 = __builtin_amdgcn_wmma_f32_16x16x32_bf16(false, (F).a, false, (F).b3, (short)0, acc3, false, false)

__global__ __launch_bounds__(256)
void sae_encode_gemm(const bf16* __restrict__ X,     // [BATCH, D_MODEL]
                     const bf16* __restrict__ W,     // [D_SAE, D_MODEL]
                     const float* __restrict__ bias, // [D_SAE]
                     float* __restrict__ pre)        // [BATCH, D_SAE]
{
    const int lane  = threadIdx.x & 31;
    const int wave  = threadIdx.x >> 5;     // 0..7
    const int m0    = blockIdx.y * 64 + (wave & 3) * 16;
    const int n0    = blockIdx.x * 128 + (wave >> 2) * 64;

    const int lrow  = lane & 15;
    const int khalf = (lane >> 4) * 8;      // lanes 0-15 -> K 0..7, lanes 16-31 -> K 8..15

    v8f acc0 = {}, acc1 = {}, acc2 = {}, acc3 = {};

    const bf16* xrow = X + (size_t)(m0 + lrow) * D_MODEL;
    const bf16* w0   = W + (size_t)(n0 +  0 + lrow) * D_MODEL;
    const bf16* w1   = W + (size_t)(n0 + 16 + lrow) * D_MODEL;
    const bf16* w2   = W + (size_t)(n0 + 32 + lrow) * D_MODEL;
    const bf16* w3   = W + (size_t)(n0 + 48 + lrow) * D_MODEL;

    Frags f0, f1;
    load_frags(f0, xrow, w0, w1, w2, w3, khalf);            // k = 0
    #pragma unroll 1
    for (int k = 0; k < D_MODEL; k += 64) {
        load_frags(f1, xrow, w0, w1, w2, w3, k + 32 + khalf);   // prefetch k+32
        WMMA4(f0);                                              // waits exclude f1 loads
        if (k + 64 < D_MODEL)
            load_frags(f0, xrow, w0, w1, w2, w3, k + 64 + khalf); // prefetch k+64
        WMMA4(f1);                                              // waits exclude f0 loads
    }

    // C/D layout: VGPR r -> M = r (lanes 0-15) / M = 8+r (lanes 16-31); N = lane&15
    const int ncol    = lane & 15;
    const int rowbase = m0 + ((lane >> 4) << 3);
    const int c0 = n0 + 0 + ncol, c1 = n0 + 16 + ncol, c2 = n0 + 32 + ncol, c3 = n0 + 48 + ncol;
    const float bv0 = bias[c0], bv1 = bias[c1], bv2 = bias[c2], bv3 = bias[c3];
    #pragma unroll
    for (int r = 0; r < 8; ++r) {
        size_t rb = (size_t)(rowbase + r) * D_SAE;
        pre[rb + c0] = acc0[r] + bv0;
        pre[rb + c1] = acc1[r] + bv1;
        pre[rb + c2] = acc2[r] + bv2;
        pre[rb + c3] = acc3[r] + bv3;
    }
}

// ------------------------- W_dec transpose ---------------------------------
// W_dec [D_MODEL, D_SAE] -> W_decT [D_SAE, D_MODEL]
__global__ __launch_bounds__(256)
void transpose_kernel(const float* __restrict__ in, float* __restrict__ out) {
    __shared__ float tile[32][33];
    int s = blockIdx.x * 32 + threadIdx.x;   // along D_SAE
    int d = blockIdx.y * 32 + threadIdx.y;   // along D_MODEL
    #pragma unroll
    for (int j = 0; j < 32; j += 8)
        tile[threadIdx.y + j][threadIdx.x] = in[(size_t)(d + j) * D_SAE + s];
    __syncthreads();
    int so = blockIdx.x * 32 + threadIdx.y;
    int dd = blockIdx.y * 32 + threadIdx.x;
    #pragma unroll
    for (int j = 0; j < 32; j += 8)
        out[(size_t)(so + j) * D_MODEL + dd] = tile[threadIdx.x][threadIdx.y + j];
}

// --------------------------- top-k radix select ----------------------------
__device__ __forceinline__ unsigned flip_f32(float f) {
    unsigned u = __float_as_uint(f);
    return (u & 0x80000000u) ? ~u : (u | 0x80000000u);   // monotonic: bigger float -> bigger key
}

__global__ __launch_bounds__(256)
void topk_kernel(float* __restrict__ latents,   // in: pre, out: sparse latents (in-place)
                 float* __restrict__ vals,      // [BATCH, KTOP]
                 int*   __restrict__ idxs)      // [BATCH, KTOP]
{
    const int row = blockIdx.x;
    const int tid = threadIdx.x;                 // 256 threads
    float* rowp = latents + (size_t)row * D_SAE;

    extern __shared__ float srow[];              // D_SAE floats = 96 KB (dynamic LDS)
    __shared__ unsigned hist[256];
    __shared__ unsigned s_prefix, s_remaining, s_tiecnt;
    __shared__ float s_v[KTOP];
    __shared__ int   s_i[KTOP];
    __shared__ unsigned s_outslot;

    for (int i = tid; i < D_SAE; i += 256) srow[i] = rowp[i];
    if (tid == 0) { s_prefix = 0; s_remaining = KTOP; s_tiecnt = 0; s_outslot = 0; }
    __syncthreads();

    // 4 MSB-first radix passes: exact key of the K-th largest element
    for (int pass = 0; pass < 4; ++pass) {
        const int shift = 24 - pass * 8;
        hist[tid] = 0;
        __syncthreads();
        const unsigned pfx = s_prefix;
        for (int i = tid; i < D_SAE; i += 256) {
            unsigned key = flip_f32(srow[i]);
            unsigned hi  = (pass == 0) ? 0u : (key >> (shift + 8));
            if (hi == pfx) atomicAdd(&hist[(key >> shift) & 255u], 1u);
        }
        __syncthreads();
        if (tid == 0) {
            unsigned rem = s_remaining, cum = 0;
            int b = 255;
            for (; b > 0; --b) { cum += hist[b]; if (cum >= rem) break; }
            if (cum < rem) cum += hist[0];               // b == 0 fallthrough
            s_remaining = rem - (cum - hist[b]);         // minus strictly-greater count
            s_prefix    = (pfx << 8) | (unsigned)b;
        }
        __syncthreads();
    }

    const unsigned thresh = s_prefix;      // full 32-bit key of K-th largest
    const unsigned rem    = s_remaining;   // how many ties at threshold to keep

    // Zero everything below threshold; collect exactly 32 (val, idx) pairs in LDS.
    for (int i = tid; i < D_SAE; i += 256) {
        float v = srow[i];
        unsigned key = flip_f32(v);
        bool keep = (key > thresh);
        if (!keep && key == thresh) {
            unsigned t = atomicAdd(&s_tiecnt, 1u);
            keep = (t < rem);
        }
        rowp[i] = keep ? v : 0.0f;
        if (keep) {
            unsigned slot = atomicAdd(&s_outslot, 1u);
            if (slot < KTOP) { s_v[slot] = v; s_i[slot] = i; }
        }
    }
    __syncthreads();

    // Deterministic decode order: sort the 32 pairs by index (tiny insertion sort).
    if (tid == 0) {
        for (int a = 1; a < KTOP; ++a) {
            float v = s_v[a]; int id = s_i[a];
            int b = a - 1;
            while (b >= 0 && s_i[b] > id) { s_i[b + 1] = s_i[b]; s_v[b + 1] = s_v[b]; --b; }
            s_i[b + 1] = id; s_v[b + 1] = v;
        }
    }
    __syncthreads();
    if (tid < KTOP) {
        vals[row * KTOP + tid] = s_v[tid];
        idxs[row * KTOP + tid] = s_i[tid];
    }
}

// ------------------------------ decode -------------------------------------
// recon[b,:] = sum_k vals[b,k] * W_decT[idx[b,k], :]
__global__ __launch_bounds__(256)
void decode_kernel(const float* __restrict__ vals, const int* __restrict__ idxs,
                   const float* __restrict__ WdecT, float* __restrict__ recon,
                   float* __restrict__ aux)
{
    const int row = blockIdx.x;
    const int tid = threadIdx.x;                 // 256 threads, 3 outputs each
    __shared__ float sv[KTOP];
    __shared__ int   si[KTOP];
    if (tid < KTOP) { sv[tid] = vals[row * KTOP + tid]; si[tid] = idxs[row * KTOP + tid]; }
    __syncthreads();

    float a0 = 0.f, a1 = 0.f, a2 = 0.f;
    #pragma unroll 4
    for (int k = 0; k < KTOP; ++k) {
        const float* col = WdecT + (size_t)si[k] * D_MODEL;
        const float v = sv[k];
        a0 += v * col[tid];
        a1 += v * col[tid + 256];
        a2 += v * col[tid + 512];
    }
    float* r = recon + (size_t)row * D_MODEL;
    r[tid] = a0; r[tid + 256] = a1; r[tid + 512] = a2;
    if (row == 0 && tid == 0) aux[0] = 0.0f;
}

// ------------------------------ launcher -----------------------------------
extern "C" void kernel_launch(void* const* d_in, const int* in_sizes, int n_in,
                              void* d_out, int out_size, void* d_ws, size_t ws_size,
                              hipStream_t stream) {
    const float* x     = (const float*)d_in[0];   // [4096, 768]
    const float* W_enc = (const float*)d_in[1];   // [24576, 768]
    const float* b_enc = (const float*)d_in[2];   // [24576]
    const float* W_dec = (const float*)d_in[3];   // [768, 24576]

    float* recon   = (float*)d_out;                          // 4096*768
    float* latents = recon + (size_t)BATCH * D_MODEL;        // 4096*24576
    float* aux     = latents + (size_t)BATCH * D_SAE;        // 1

    char* ws = (char*)d_ws;
    size_t off = 0;
    bf16*  x_bf   = (bf16*)(ws + off);  off += (size_t)BATCH * D_MODEL * sizeof(bf16);   // 6.3 MB
    off = (off + 255) & ~(size_t)255;
    bf16*  w_bf   = (bf16*)(ws + off);  off += (size_t)D_SAE * D_MODEL * sizeof(bf16);   // 37.7 MB
    off = (off + 255) & ~(size_t)255;
    float* wdecT  = (float*)(ws + off); off += (size_t)D_SAE * D_MODEL * sizeof(float);  // 75.5 MB
    off = (off + 255) & ~(size_t)255;
    float* vals   = (float*)(ws + off); off += (size_t)BATCH * KTOP * sizeof(float);     // 0.5 MB
    off = (off + 255) & ~(size_t)255;
    int*   idx    = (int*)(ws + off);                                                     // 0.5 MB

    // 1) fp32 -> bf16 operand conversion
    {
        int n8 = BATCH * D_MODEL / 8;
        cvt_bf16_kernel<<<(n8 + 255) / 256, 256, 0, stream>>>(x, x_bf, n8);
        n8 = D_SAE * D_MODEL / 8;
        cvt_bf16_kernel<<<(n8 + 255) / 256, 256, 0, stream>>>(W_enc, w_bf, n8);
    }

    // 2) transpose W_dec for coalesced sparse decode
    transpose_kernel<<<dim3(D_SAE / 32, D_MODEL / 32), dim3(32, 8), 0, stream>>>(W_dec, wdecT);

    // 3) encode GEMM: pre -> latents region of d_out (bias fused)
    sae_encode_gemm<<<dim3(D_SAE / 128, BATCH / 64), 256, 0, stream>>>(x_bf, w_bf, b_enc, latents);

    // 4) exact top-32 per row, in place; 96 KB dynamic LDS per block (WGP has 320 KB)
    topk_kernel<<<BATCH, 256, D_SAE * sizeof(float), stream>>>(latents, vals, idx);

    // 5) sparse decode + aux_loss = 0
    decode_kernel<<<BATCH, 256, 0, stream>>>(vals, idx, wdecT, recon, aux);
}
